// SingleHeadAttention_72885595013606
// MI455X (gfx1250) — matrix operand order repaired and checked by reading, not compiled, not run on hardware
//
#include <hip/hip_runtime.h>
#include <hip/hip_bf16.h>
#include <math.h>

// ---------------------------------------------------------------------------
// SingleHeadAttention for gfx1250 (MI455X), wave32 + WMMA f16->f32.
//
//   qp = q@WQ^T+bq ; kp = k@WK^T+bk ; vp = v@WV^T+bv
//   E[b,q,k]   = (q>=k) ? exp(scale * qp[q]·kp[k]) : 1   (mult. tril mask)
//   colsum[b,k]= sum_q E[b,q,k]                          (softmax over axis=1)
//   vT[b,d,k]  = vp[b,k,d] / colsum[b,k]                 (fold softmax into V)
//   out[b,q,d] = sum_k E[b,q,k] * vT[b,d,k]
//
// GEMMs: 256 threads / 8 waves, 128x128 block tile, 32x64 wave tile,
// BK=32, double-buffered LDS; f16 operand staging via async global->LDS
// DMA (ASYNCcnt) when available, overlapped with v_wmma_f32_16x16x32_f16.
// ---------------------------------------------------------------------------

typedef __attribute__((ext_vector_type(16))) _Float16 v16h;
typedef __attribute__((ext_vector_type(8)))  _Float16 v8h;
typedef __attribute__((ext_vector_type(8)))  float    v8f;
typedef int v4i __attribute__((vector_size(16)));

#define WMMA16(a, b, c) \
  __builtin_amdgcn_wmma_f32_16x16x32_f16(false, (a), false, (b), (short)0, (c), false, false)

#if defined(__gfx1250__) && __has_builtin(__builtin_amdgcn_global_load_async_to_lds_b128)
#define USE_ASYNC_LDS 1
#else
#define USE_ASYNC_LDS 0
#endif

static constexpr int   kB = 4, kS = 2048, kD = 512;
static constexpr float kScale = 0.044194173824159216f; // 1/sqrt(512)

// Block geometry
static constexpr int BM = 128, BN = 128, BK = 32, NTHREADS = 256;
static constexpr int TILE_HALVES = BM * BK; // 4096 halves = 8KB per operand tile

__device__ __forceinline__ void wait_async() {
#if USE_ASYNC_LDS
#if __has_builtin(__builtin_amdgcn_s_wait_asynccnt)
  __builtin_amdgcn_s_wait_asynccnt(0);
#else
  asm volatile("s_wait_asynccnt 0x0" ::: "memory");
#endif
#endif
}

// ---- LDS tile staging -------------------------------------------------------
// LDS tile: [128 rows][32 halves], row stride 64 B.

template <int LD>
__device__ __forceinline__ void stage16(_Float16* dst, const _Float16* src, int tid) {
  // 128 rows * 4 chunks of 16B = 512 chunks, 256 threads -> 2 each
#pragma unroll
  for (int j = 0; j < 2; j++) {
    const int c = tid + j * NTHREADS;
    const int row = c >> 2, off = (c & 3) << 3;
    *(v8h*)(dst + row * BK + off) = *(const v8h*)(src + (size_t)row * LD + off);
  }
}

#if USE_ASYNC_LDS
template <int LD>
__device__ __forceinline__ void stage_async(_Float16* dst, const _Float16* src, int tid) {
#pragma unroll
  for (int j = 0; j < 2; j++) {
    const int c = tid + j * NTHREADS;
    const int row = c >> 2, off = (c & 3) << 3;
    __builtin_amdgcn_global_load_async_to_lds_b128(
        (__attribute__((address_space(1))) v4i*)(void*)(src + (size_t)row * LD + off),
        (__attribute__((address_space(3))) v4i*)(void*)(dst + row * BK + off),
        0, 0);
  }
}
#endif

template <int LD>
__device__ __forceinline__ void stage32(_Float16* dst, const float* src, int tid) {
  // 128 rows * 8 chunks of 4 floats = 1024 chunks, convert f32 -> f16
#pragma unroll
  for (int j = 0; j < 4; j++) {
    const int c = tid + j * NTHREADS;
    const int row = c >> 3, off = (c & 7) << 2;
    const float4 v = *(const float4*)(src + (size_t)row * LD + off);
    _Float16* d = dst + row * BK + off;
    d[0] = (_Float16)v.x; d[1] = (_Float16)v.y;
    d[2] = (_Float16)v.z; d[3] = (_Float16)v.w;
  }
}

// ---- WMMA fragment loads (layouts per CDNA5 ISA 7.12.2) ----------

// A 16x32 f16: lane = (row, K-half); interleaved K pairs per ISA A-layout.
__device__ __forceinline__ v16h load_fragA(const _Float16* tile, int rowBase, int lane) {
  const int r = lane & 15, half = lane >> 4;
  const _Float16* p = tile + (rowBase + r) * BK;
  const v8h lo = *(const v8h*)(p + half * 8);       // halves 0..7 : K = half*8..
  const v8h hi = *(const v8h*)(p + 16 + half * 8);  // halves 8..15: K = 16+half*8..
  return __builtin_shufflevector(lo, hi, 0, 1, 2, 3, 4, 5, 6, 7,
                                 8, 9, 10, 11, 12, 13, 14, 15);
}

// B 32x16 f16: lane = (col, K-half); interleaved K groups matching the A layout.
__device__ __forceinline__ v16h load_fragB(const _Float16* tile, int colBase, int lane) {
  const int n = lane & 15, half = lane >> 4;
  const _Float16* p = tile + (colBase + n) * BK;
  const v8h lo = *(const v8h*)(p + half * 8);
  const v8h hi = *(const v8h*)(p + 16 + half * 8);
  return __builtin_shufflevector(lo, hi, 0, 1, 2, 3, 4, 5, 6, 7,
                                 8, 9, 10, 11, 12, 13, 14, 15);
}

// ---- block GEMM core: C(128x128) += A(128xK) * Bt(128xK)^T ------------------
// A, B point at (rowBase, 0) / (colBase, 0). smem: 2 x (Atile+Btile) halves.

template <int LDA, int LDB, int KTOT, bool F32SRC>
__device__ __forceinline__ void gemm_core(const void* A, const void* B,
                                          _Float16* smem, v8f acc[2][4]) {
  const int tid  = threadIdx.x;
  const int lane = tid & 31;
  const int wave = tid >> 5;
  const int wm = wave & 3;   // 4 row groups of 32
  const int wn = wave >> 2;  // 2 col groups of 64
  constexpr int NSTEP = KTOT / BK;

  auto stage = [&](int buf, int k0) {
    _Float16* dA = smem + buf * (2 * TILE_HALVES);
    _Float16* dB = dA + TILE_HALVES;
    if (F32SRC) {
      stage32<LDA>(dA, (const float*)A + k0, tid);
      stage32<LDB>(dB, (const float*)B + k0, tid);
    } else {
#if USE_ASYNC_LDS
      stage_async<LDA>(dA, (const _Float16*)A + k0, tid);
      stage_async<LDB>(dB, (const _Float16*)B + k0, tid);
#else
      stage16<LDA>(dA, (const _Float16*)A + k0, tid);
      stage16<LDB>(dB, (const _Float16*)B + k0, tid);
#endif
    }
  };

  stage(0, 0);
  if (!F32SRC) wait_async();
  __syncthreads();

  for (int i = 0; i < NSTEP; i++) {
    const int cur = i & 1;
    if (i + 1 < NSTEP) stage(cur ^ 1, (i + 1) * BK);  // DMA next tile

    const _Float16* lA = smem + cur * (2 * TILE_HALVES);
    const _Float16* lB = lA + TILE_HALVES;
    const v16h a0 = load_fragA(lA, wm * 32, lane);
    const v16h a1 = load_fragA(lA, wm * 32 + 16, lane);
    v16h bf[4];
#pragma unroll
    for (int j = 0; j < 4; j++) bf[j] = load_fragB(lB, wn * 64 + j * 16, lane);
#pragma unroll
    for (int j = 0; j < 4; j++) {
      acc[0][j] = WMMA16(a0, bf[j], acc[0][j]);
      acc[1][j] = WMMA16(a1, bf[j], acc[1][j]);
    }

    if (!F32SRC) wait_async();  // next tile's DMA done (overlapped with WMMA)
    __syncthreads();
  }
}

// Epilogue walker: f(m, n, value) per owned C element.
template <typename F>
__device__ __forceinline__ void store_tile(v8f acc[2][4], int rowBase, int colBase, F&& f) {
  const int lane = threadIdx.x & 31, wave = threadIdx.x >> 5;
  const int wm = wave & 3, wn = wave >> 2;
#pragma unroll
  for (int ti = 0; ti < 2; ti++)
#pragma unroll
    for (int j = 0; j < 4; j++) {
      const int n = colBase + wn * 64 + j * 16 + (lane & 15);
#pragma unroll
      for (int r = 0; r < 8; r++) {
        const int m = rowBase + wm * 32 + ti * 16 + (lane >> 4) * 8 + r;
        f(m, n, acc[ti][j][r]);
      }
    }
}

// ---- kernels ----------------------------------------------------------------

// out = X @ W^T + bias ; X:[M,512] f32, W:[512,512] f32 ([N,K] row-major)
__global__ __launch_bounds__(NTHREADS) void proj_kernel(
    const float* __restrict__ X, const float* __restrict__ W,
    const float* __restrict__ bias, _Float16* __restrict__ outH,
    float* __restrict__ outF) {
  __shared__ _Float16 smem[4 * TILE_HALVES];
  const int rowBase = blockIdx.x * BM, colBase = blockIdx.y * BN;
  v8f acc[2][4] = {};
  gemm_core<kD, kD, kD, true>(X + (size_t)rowBase * kD, W + (size_t)colBase * kD,
                              smem, acc);
  store_tile(acc, rowBase, colBase, [&](int m, int n, float val) {
    val += bias[n];
    const size_t o = (size_t)m * kD + n;
    if (outH) outH[o] = (_Float16)val;
    if (outF) outF[o] = val;
  });
}

// E[b,q,k] = (q>=k) ? exp(scale * qp[q]·kp[k]) : exp(0)==1
__global__ __launch_bounds__(NTHREADS) void qk_kernel(
    const _Float16* __restrict__ QP, const _Float16* __restrict__ KP,
    _Float16* __restrict__ E) {
  __shared__ _Float16 smem[4 * TILE_HALVES];
  const int b = blockIdx.z;
  const int rowBase = blockIdx.x * BM, colBase = blockIdx.y * BN;
  const _Float16* A = QP + (size_t)b * kS * kD + (size_t)rowBase * kD;
  const _Float16* B = KP + (size_t)b * kS * kD + (size_t)colBase * kD;
  v8f acc[2][4] = {};
  gemm_core<kD, kD, kD, false>(A, B, smem, acc);
  _Float16* Eb = E + (size_t)b * kS * kS;
  store_tile(acc, rowBase, colBase, [&](int m, int n, float val) {
    const float e = (m >= n) ? __expf(val * kScale) : 1.0f;
    Eb[(size_t)m * kS + n] = (_Float16)e;
  });
}

// colsum[b,k] = sum_q E[b,q,k]   (softmax over the query axis)
__global__ __launch_bounds__(256) void colsum_kernel(
    const _Float16* __restrict__ E, float* __restrict__ colsum) {
  const int b = blockIdx.y;
  const int k = blockIdx.x * 256 + threadIdx.x;
  const _Float16* p = E + (size_t)b * kS * kS + k;
  float s = 0.0f;
  for (int q = 0; q < kS; q++) s += (float)p[(size_t)q * kS];
  colsum[b * kS + k] = s;
}

// vT[b,d,k] = f16( vp[b,k,d] / colsum[b,k] )   (transposed for GEMM B operand)
__global__ __launch_bounds__(256) void vscale_kernel(
    const float* __restrict__ vp, const float* __restrict__ colsum,
    _Float16* __restrict__ vT) {
  const int b = blockIdx.z;
  const int idx = blockIdx.x * 256 + threadIdx.x;  // over 512*2048
  const int d = idx >> 11, k = idx & (kS - 1);
  const float inv = 1.0f / colsum[b * kS + k];
  vT[(size_t)b * kD * kS + idx] =
      (_Float16)(vp[((size_t)b * kS + k) * kD + d] * inv);
}

// out[b,q,d] = sum_k E[b,q,k] * vT[b,d,k]
__global__ __launch_bounds__(NTHREADS) void av_kernel(
    const _Float16* __restrict__ E, const _Float16* __restrict__ VT,
    float* __restrict__ OUT) {
  __shared__ _Float16 smem[4 * TILE_HALVES];
  const int b = blockIdx.z;
  const int rowBase = blockIdx.x * BM, colBase = blockIdx.y * BN;
  const _Float16* A = E + (size_t)b * kS * kS + (size_t)rowBase * kS;
  const _Float16* B = VT + (size_t)b * kD * kS + (size_t)colBase * kS;
  v8f acc[2][4] = {};
  gemm_core<kS, kS, kS, false>(A, B, smem, acc);
  float* Ob = OUT + (size_t)b * kS * kD;
  store_tile(acc, rowBase, colBase, [&](int m, int n, float val) {
    Ob[(size_t)m * kD + n] = val;
  });
}

// ---- launch -----------------------------------------------------------------

extern "C" void kernel_launch(void* const* d_in, const int* in_sizes, int n_in,
                              void* d_out, int out_size, void* d_ws,
                              size_t ws_size, hipStream_t stream) {
  (void)in_sizes; (void)n_in; (void)out_size; (void)ws_size;
  const float* q   = (const float*)d_in[0];
  const float* k   = (const float*)d_in[1];
  const float* v   = (const float*)d_in[2];
  const float* WQw = (const float*)d_in[3];
  const float* WQb = (const float*)d_in[4];
  const float* WKw = (const float*)d_in[5];
  const float* WKb = (const float*)d_in[6];
  const float* WVw = (const float*)d_in[7];
  const float* WVb = (const float*)d_in[8];
  float* out = (float*)d_out;

  // workspace carve-up (~75.6 MB total)
  char* ws = (char*)d_ws;
  size_t off = 0;
  auto take = [&](size_t bytes) {
    char* p = ws + off;
    off += (bytes + 255) & ~(size_t)255;
    return p;
  };
  _Float16* qpH  = (_Float16*)take((size_t)kB * kS * kD * 2);
  _Float16* kpH  = (_Float16*)take((size_t)kB * kS * kD * 2);
  float*    vpF  = (float*)   take((size_t)kB * kS * kD * 4);
  _Float16* E    = (_Float16*)take((size_t)kB * kS * kS * 2);
  float*    csum = (float*)   take((size_t)kB * kS * 4);
  _Float16* vT   = (_Float16*)take((size_t)kB * kD * kS * 2);

  const dim3 blk(NTHREADS);
  const dim3 gProj((kB * kS) / BM, kD / BN, 1);
  proj_kernel<<<gProj, blk, 0, stream>>>(q, WQw, WQb, qpH, nullptr);
  proj_kernel<<<gProj, blk, 0, stream>>>(k, WKw, WKb, kpH, nullptr);
  proj_kernel<<<gProj, blk, 0, stream>>>(v, WVw, WVb, nullptr, vpF);

  const dim3 gQK(kS / BM, kS / BN, kB);
  qk_kernel<<<gQK, blk, 0, stream>>>(qpH, kpH, E);

  colsum_kernel<<<dim3(kS / 256, kB), 256, 0, stream>>>(E, csum);
  vscale_kernel<<<dim3((kD * kS) / 256, 1, kB), 256, 0, stream>>>(vpF, csum, vT);

  const dim3 gAV(kS / BM, kD / BN, kB);
  av_kernel<<<gAV, blk, 0, stream>>>(E, vT, out);
}
